// GINLayer_7000796693167
// MI455X (gfx1250) — matrix operand order repaired and checked by reading, not compiled, not run on hardware
//
#include <hip/hip_runtime.h>

// ---------------------------------------------------------------------------
// GIN layer for MI455X (gfx1250, wave32, WMMA).
//   h   = x + segment_sum(x[src] -> dst)            (fp32, L2 atomics)
//   h1  = relu(h @ W1^T + b1)                       (bf16-in WMMA, f32 acc)
//   out = relu(h1 @ W2^T + b2)                      (bf16-in WMMA, f32 acc)
// D fixed at 128 (4 K-steps of v_wmma_f32_16x16x32_bf16).
// Both GEMMs are bandwidth-bound (6.6 GFLOP vs ~150 MB traffic @ 23.3 TB/s),
// so: bf16 operands halve input traffic, 64-row blocks give 4x B-fragment
// reuse, and the epilogue goes through LDS for coalesced b128 stores.
// ---------------------------------------------------------------------------

typedef __attribute__((ext_vector_type(16))) __bf16         bf16x16;
typedef __attribute__((ext_vector_type(8)))  float          f32x8;
typedef __attribute__((ext_vector_type(4)))  float          f32x4;
typedef __attribute__((ext_vector_type(8)))  unsigned short u16x8;
typedef __attribute__((ext_vector_type(16))) unsigned short u16x16;
typedef __attribute__((ext_vector_type(4)))  int            i32x4;

#if __has_builtin(__builtin_amdgcn_global_load_async_to_lds_b128) && \
    __has_builtin(__builtin_amdgcn_s_wait_asynccnt)
#define GIN_HAVE_ASYNC_LDS 1
#else
#define GIN_HAVE_ASYNC_LDS 0
#endif

union Frag { u16x16 v; u16x8 h[2]; };

__device__ __forceinline__ unsigned short f32_to_bf16(float f) {
  // round-to-nearest-even
  unsigned int u = __builtin_bit_cast(unsigned int, f);
  u += 0x7FFFu + ((u >> 16) & 1u);
  return (unsigned short)(u >> 16);
}

// ---------------------------------------------------------------------------
// h = x  (the (1+eps)*x term with eps==0), float4 copy
// ---------------------------------------------------------------------------
__global__ __launch_bounds__(256) void gin_init_h(const f32x4* __restrict__ x,
                                                  f32x4* __restrict__ h, int nvec) {
  int t = blockIdx.x * 256 + threadIdx.x;
  if (t < nvec) h[t] = x[t];
}

// ---------------------------------------------------------------------------
// Scatter-add: one wave per edge; lane l handles features [4l, 4l+4).
// Coalesced 512B gather of the source row, fp32 global atomics into h[dst].
// ---------------------------------------------------------------------------
__global__ __launch_bounds__(256) void gin_scatter(const float* __restrict__ x,
                                                   const int* __restrict__ src,
                                                   const int* __restrict__ dst,
                                                   float* __restrict__ h, int E) {
  int wid  = (blockIdx.x * 256 + threadIdx.x) >> 5;
  int lane = threadIdx.x & 31;
  if (wid >= E) return;
  int s = src[wid];
  int d = dst[wid];
  f32x4 v = ((const f32x4*)(x + (size_t)s * 128))[lane];
  float* hp = h + (size_t)d * 128 + lane * 4;
  atomicAdd(hp + 0, v[0]);
  atomicAdd(hp + 1, v[1]);
  atomicAdd(hp + 2, v[2]);
  atomicAdd(hp + 3, v[3]);
}

// ---------------------------------------------------------------------------
// Convert both weight matrices fp32 -> bf16 (row-major W[n][k], 128x128 each).
// 64 KB total -> L2 resident for every GEMM block afterwards.
// ---------------------------------------------------------------------------
__global__ __launch_bounds__(256) void gin_cvt_w(const float* __restrict__ W1,
                                                 const float* __restrict__ W2,
                                                 unsigned short* __restrict__ W1b,
                                                 unsigned short* __restrict__ W2b,
                                                 int n) {
  int t = blockIdx.x * 256 + threadIdx.x;
  if (t < n)          W1b[t]     = f32_to_bf16(W1[t]);
  else if (t < 2 * n) W2b[t - n] = f32_to_bf16(W2[t - n]);
}

// ---------------------------------------------------------------------------
// Fused GEMM + bias + ReLU:  Out[RB x 128] = act(A[RB x 128] @ W^T + b)
// Block = 256 threads = 8 waves; wave w owns output cols [16w, 16w+16).
// RB = 64 rows per block -> each wave runs 4 M-subtiles, reusing each
// B fragment 4x. A tile staged in LDS as bf16 (16 KB); C staged in the same
// smem after a barrier for coalesced b128 global stores (<= 32 KB).
//
// ISA fragment layouts (CDNA5 ISA):
//   A 16x32 bf16 : lanes 0-15 -> M=lane,  elems 0..7 = K k0..k0+7,
//                                elems 8..15 = K k0+16..k0+23  (k0 = hi ? 8 : 0)
//   B 32x16 bf16 : lane = column, elems 0..7 = K k0..k0+7, elems 8..15 = K k0+16..k0+23 (as A)
//   C 16x16 f32  : elem r -> row r + (hi?8:0), col = lane&15
//
// Staging is deliberately unguarded: the last partial tile over-reads into
// the next workspace region (still inside d_ws), and a garbage A-row only
// affects the matching (unstored) C-row. Only the flush checks row < rows.
// ---------------------------------------------------------------------------
constexpr int RB = 64;

template <bool IN_BF16, bool OUT_BF16>
__global__ __launch_bounds__(256) void gin_gemm(const void* __restrict__ Hin,
                                                const unsigned short* __restrict__ Wb,
                                                const float* __restrict__ bias,
                                                void* __restrict__ Outp, int N) {
  __shared__ __align__(16) char smem[RB * 128 * 4];  // 32 KB (A: first 16 KB)
  unsigned short* ldsA = (unsigned short*)smem;
  const int tid  = threadIdx.x;
  const int row0 = blockIdx.x * RB;
  int rows = N - row0;
  if (rows > RB) rows = RB;

  // ---- stage A tile (RB x 128 bf16): thread t owns elems [32t, 32t+32) ----
  if constexpr (IN_BF16) {
    const unsigned short* g = (const unsigned short*)Hin + (size_t)row0 * 128 + tid * 32;
    unsigned short* l = ldsA + tid * 32;
#if GIN_HAVE_ASYNC_LDS
#pragma unroll
    for (int i = 0; i < 4; ++i)
      __builtin_amdgcn_global_load_async_to_lds_b128(
          (i32x4*)(g + i * 8), (i32x4*)(l + i * 8), 0, 0);
    __builtin_amdgcn_s_wait_asynccnt(0);
#else
#pragma unroll
    for (int i = 0; i < 4; ++i)
      *(u16x8*)(l + i * 8) = *(const u16x8*)(g + i * 8);
#endif
  } else {
    const float* Hf = (const float*)Hin;
    const f32x4* p = (const f32x4*)(Hf + (size_t)row0 * 128 + tid * 32);
#pragma unroll
    for (int c = 0; c < 4; ++c) {
      f32x4 v0 = p[2 * c], v1 = p[2 * c + 1];
      u16x8 o;
#pragma unroll
      for (int i = 0; i < 4; ++i) { o[i] = f32_to_bf16(v0[i]); o[4 + i] = f32_to_bf16(v1[i]); }
      *(u16x8*)(ldsA + tid * 32 + c * 8) = o;
    }
  }
  __syncthreads();

  // ---- WMMA main loop: 4 K-chunks x 4 M-subtiles ----
  const int lane = tid & 31, wv = tid >> 5;
  const int hi = lane >> 4, ln = lane & 15;
  const int col = (wv << 4) + ln;
  const int rb  = hi << 3;

  f32x8 acc[4] = {};
#pragma unroll
  for (int kc = 0; kc < 4; ++kc) {
    const int k0 = kc << 5;
    Frag b;
    const unsigned short* bp = Wb + col * 128 + k0 + (hi << 3);
    b.h[0] = *(const u16x8*)(bp);
    b.h[1] = *(const u16x8*)(bp + 16);
    const bf16x16 bv = __builtin_bit_cast(bf16x16, b.v);
#pragma unroll
    for (int mt = 0; mt < 4; ++mt) {
      Frag a;
      const unsigned short* ap = ldsA + (mt * 16 + ln) * 128 + k0 + (hi << 3);
      a.h[0] = *(const u16x8*)(ap);
      a.h[1] = *(const u16x8*)(ap + 16);
      acc[mt] = __builtin_amdgcn_wmma_f32_16x16x32_bf16(
          false, __builtin_bit_cast(bf16x16, a.v), false, bv,
          (short)0, acc[mt], false, false);
    }
  }

  // ---- epilogue: bias + ReLU, stage C in LDS, coalesced flush ----
  __syncthreads();  // all A reads complete before smem is reused for C
  const float bv = bias[col];
  if constexpr (OUT_BF16) {
    unsigned short* ldsC = (unsigned short*)smem;
#pragma unroll
    for (int mt = 0; mt < 4; ++mt)
#pragma unroll
      for (int r = 0; r < 8; ++r) {
        float v = acc[mt][r] + bv;
        v = v > 0.f ? v : 0.f;
        ldsC[(mt * 16 + rb + r) * 128 + col] = f32_to_bf16(v);
      }
  } else {
    float* ldsC = (float*)smem;
#pragma unroll
    for (int mt = 0; mt < 4; ++mt)
#pragma unroll
      for (int r = 0; r < 8; ++r) {
        float v = acc[mt][r] + bv;
        ldsC[(mt * 16 + rb + r) * 128 + col] = v > 0.f ? v : 0.f;
      }
  }
  __syncthreads();

  const int frow = tid >> 2;  // 4 threads per output row in the flush
  if (frow < rows) {
    if constexpr (OUT_BF16) {
      const u16x8* s = (const u16x8*)((const unsigned short*)smem + tid * 32);
      u16x8* d = (u16x8*)((unsigned short*)Outp + (size_t)row0 * 128 + tid * 32);
#pragma unroll
      for (int i = 0; i < 4; ++i) d[i] = s[i];
    } else {
      const f32x4* s = (const f32x4*)((const float*)smem + tid * 32);
      f32x4* d = (f32x4*)((float*)Outp + (size_t)row0 * 128 + tid * 32);
#pragma unroll
      for (int i = 0; i < 8; ++i) d[i] = s[i];
    }
  }
}

// ---------------------------------------------------------------------------
// Launch. Workspace layout (256B-aligned blocks):
//   [0, N*D*4)                    : h   fp32
//   [N*D*4, +N*D*2)               : h1  bf16
//   then W1b (D*D*2), W2b (D*D*2) : bf16 weights
// ~76.9 MB for N=100000, D=128. Last-tile staging over-reads stay inside ws.
// ---------------------------------------------------------------------------
extern "C" void kernel_launch(void* const* d_in, const int* in_sizes, int n_in,
                              void* d_out, int out_size, void* d_ws, size_t ws_size,
                              hipStream_t stream) {
  const float* x  = (const float*)d_in[0];
  const int*   ei = (const int*)d_in[1];
  const float* W1 = (const float*)d_in[2];
  const float* b1 = (const float*)d_in[3];
  const float* W2 = (const float*)d_in[4];
  const float* b2 = (const float*)d_in[5];
  float* out = (float*)d_out;

  const int D = 128;
  const int N = in_sizes[0] / D;
  const int E = in_sizes[1] / 2;

  char* ws = (char*)d_ws;
  float*          h   = (float*)ws;
  const size_t    hB  = (size_t)N * D * sizeof(float);
  unsigned short* h1  = (unsigned short*)(ws + hB);
  const size_t    h1B = (size_t)N * D * sizeof(unsigned short);
  unsigned short* W1b = (unsigned short*)(ws + hB + h1B);
  unsigned short* W2b = W1b + D * D;

  // 1) bf16 weight conversion
  gin_cvt_w<<<(2 * D * D + 255) / 256, 256, 0, stream>>>(W1, W2, W1b, W2b, D * D);

  // 2) h = x
  const int nvec = N * D / 4;
  gin_init_h<<<(nvec + 255) / 256, 256, 0, stream>>>((const f32x4*)x, (f32x4*)h, nvec);

  // 3) scatter-add: one wave per edge (8 edges per 256-thread block)
  gin_scatter<<<(E + 7) / 8, 256, 0, stream>>>(x, ei, ei + E, h, E);

  // 4) MLP: Linear+ReLU (bf16 out), then Linear+ReLU (fp32 out)
  const int mtiles = (N + RB - 1) / RB;
  gin_gemm<false, true><<<mtiles, 256, 0, stream>>>(h, W1b, b1, h1, N);
  gin_gemm<true, false><<<mtiles, 256, 0, stream>>>(h1, W2b, b2, out, N);
}
